// SphericalFourierNeuralOperatorNet_35519379538273
// MI455X (gfx1250) — matrix operand order repaired and checked by reading, not compiled, not run on hardware
//
#include <hip/hip_runtime.h>
#include <hip/hip_bf16.h>
#include <math.h>

// ---------------------------------------------------------------------------
// SFNO forward for gfx1250 (MI455X). DFT-as-matmul (H=181 is prime), all
// heavy math through v_wmma_f32_16x16x32_bf16 (f32 accumulate).
// Frequency (kw) dimension padded 181 -> 184 so every hot operand has
// 16B-aligned rows => b128 global loads + packed LDS stores in the staging.
// ---------------------------------------------------------------------------

typedef __attribute__((ext_vector_type(16))) __bf16 v16bf;
typedef __attribute__((ext_vector_type(8)))  float  v8f;

union FragU { uint4 u[2]; v16bf v; };

__device__ __forceinline__ unsigned short f32_to_bf16(float f) {
  union { float f; unsigned u; } c; c.f = f;
  unsigned u = c.u;
  return (unsigned short)((u + 0x7FFFu + ((u >> 16) & 1u)) >> 16); // RNE
}

__device__ __forceinline__ unsigned pack2bf(float lo, float hi) {
  return (unsigned)f32_to_bf16(lo) | ((unsigned)f32_to_bf16(hi) << 16);
}

__device__ __forceinline__ float gelu_exact(float x) {
  return 0.5f * x * (1.0f + erff(x * 0.70710678118654752f));
}

// ---------------------------------------------------------------------------
// GEMM: C[b] = act( alpha * op(A[b])*B[b] + beta*C[b] + bias[m] + D )
//   A: MxK f32 (lda) or KxM if transA (A[k*lda+m]);  B: KxN f32 (ldb)
//   act: 0 none, 1 relu (CReLU real part), 2 exact gelu
// Tile 64(M) x 128(N), K-step 32. 256 thr = 8 waves; wave -> 16x64 of C
// (one A frag reused by 4 B frags / 4 WMMAs per K-step).
// ---------------------------------------------------------------------------
__global__ __launch_bounds__(256) void gemm_bf16_k(
    const float* __restrict__ A, const float* __restrict__ B,
    float* __restrict__ C, const float* __restrict__ D,
    const float* __restrict__ bias,
    int M, int N, int K, int lda, int ldb, int ldc,
    long long asb, long long bsb, long long csb,
    int transA, float alpha, int beta, int act)
{
  __shared__ __align__(16) unsigned short As[64  * 40]; // [m][k], 80B row stride
  __shared__ __align__(16) unsigned short Bs[128 * 40]; // [n][k] (transposed)

  const int tid  = threadIdx.x;
  const int mblk = blockIdx.y * 64;
  const int nblk = blockIdx.x * 128;
  const long long bz = blockIdx.z;
  const float* Ab = A + bz * asb;
  const float* Bb = B + bz * bsb;
  float*       Cb = C + bz * csb;

  const int wv = tid >> 5, lane = tid & 31;
  const int mrow = wv & 3;   // 16-row strip
  const int ns   = wv >> 2;  // 0/1 -> n range [ns*64, ns*64+64)

  const bool alnA = ((lda & 3) == 0);
  const bool alnB = ((ldb & 3) == 0);
  const bool fullM = (mblk + 64 <= M);
  const bool fullN = (nblk + 128 <= N);

  v8f acc[4] = {};

  const int nk = (K + 31) >> 5;
  for (int kb = 0; kb < nk; ++kb) {
    const int k0 = kb << 5;
    const bool fullK = (k0 + 32 <= K);

    // ---------------- stage A (64 x 32) ----------------
    if (!transA) {
      if (fullM && fullK && alnA) {
        // 512 float4s, 2 per thread; b128 loads, b64 LDS stores
        #pragma unroll
        for (int j = 0; j < 2; ++j) {
          const int q = tid + 256 * j;
          const int row = q >> 3, kq = (q & 7) << 2;
          const float4 f = *(const float4*)(Ab + (long long)(mblk + row) * lda + k0 + kq);
          uint2 p; p.x = pack2bf(f.x, f.y); p.y = pack2bf(f.z, f.w);
          *(uint2*)(As + row * 40 + kq) = p;
        }
      } else {
        const int row = tid >> 5, kk = tid & 31;
        #pragma unroll
        for (int j = 0; j < 8; ++j) {
          const int rr = row + j * 8;
          const int m = mblk + rr, k = k0 + kk;
          const int mc = m < M ? m : M - 1, kc = k < K ? k : K - 1;
          float v = Ab[(long long)mc * lda + kc];
          v = (m < M && k < K) ? v : 0.0f;
          As[rr * 40 + kk] = f32_to_bf16(v);
        }
      }
    } else {
      if (fullM && fullK && alnA) {
        // k-pairs: 16 kp x 16 m-float4s = 256 -> 1 per thread
        const int kp = tid >> 4, mq = (tid & 15) << 2;
        const long long r0 = (long long)(k0 + 2 * kp) * lda + mblk + mq;
        const float4 f0 = *(const float4*)(Ab + r0);
        const float4 f1 = *(const float4*)(Ab + r0 + lda);
        *(unsigned*)(As + (mq + 0) * 40 + 2 * kp) = pack2bf(f0.x, f1.x);
        *(unsigned*)(As + (mq + 1) * 40 + 2 * kp) = pack2bf(f0.y, f1.y);
        *(unsigned*)(As + (mq + 2) * 40 + 2 * kp) = pack2bf(f0.z, f1.z);
        *(unsigned*)(As + (mq + 3) * 40 + 2 * kp) = pack2bf(f0.w, f1.w);
      } else {
        const int mcol = tid & 63, kk0 = tid >> 6;
        #pragma unroll
        for (int j = 0; j < 8; ++j) {
          const int kk = kk0 + j * 4;
          const int m = mblk + mcol, k = k0 + kk;
          const int mc = m < M ? m : M - 1, kc = k < K ? k : K - 1;
          float v = Ab[(long long)kc * lda + mc];
          v = (m < M && k < K) ? v : 0.0f;
          As[mcol * 40 + kk] = f32_to_bf16(v);
        }
      }
    }

    // ---------------- stage B (32 x 128) transposed -> Bs[n][k] ----------------
    if (fullN && fullK && alnB) {
      // k-pairs: 16 kp x 32 n-float4s = 512 -> 2 per thread
      #pragma unroll
      for (int j = 0; j < 2; ++j) {
        const int q = tid + 256 * j;
        const int kp = q >> 5, nq = (q & 31) << 2;
        const long long r0 = (long long)(k0 + 2 * kp) * ldb + nblk + nq;
        const float4 f0 = *(const float4*)(Bb + r0);
        const float4 f1 = *(const float4*)(Bb + r0 + ldb);
        *(unsigned*)(Bs + (nq + 0) * 40 + 2 * kp) = pack2bf(f0.x, f1.x);
        *(unsigned*)(Bs + (nq + 1) * 40 + 2 * kp) = pack2bf(f0.y, f1.y);
        *(unsigned*)(Bs + (nq + 2) * 40 + 2 * kp) = pack2bf(f0.z, f1.z);
        *(unsigned*)(Bs + (nq + 3) * 40 + 2 * kp) = pack2bf(f0.w, f1.w);
      }
    } else {
      const int ncol = tid & 127, kk0 = tid >> 7;
      #pragma unroll
      for (int j = 0; j < 16; ++j) {
        const int kk = kk0 + j * 2;
        const int n = nblk + ncol, k = k0 + kk;
        const int nc = n < N ? n : N - 1, kc = k < K ? k : K - 1;
        float v = Bb[(long long)kc * ldb + nc];
        v = (n < N && k < K) ? v : 0.0f;
        Bs[ncol * 40 + kk] = f32_to_bf16(v);
      }
    }
    __syncthreads();

    // ---------------- fragments + WMMA (ISA 7.12.2 16-bit layouts) ----------
    const int offA = (lane >= 16) ? 8 : 0;
    const int arow = mrow * 16 + (lane & 15);
    FragU fa;
    fa.u[0] = *(const uint4*)(As + arow * 40 + offA);
    fa.u[1] = *(const uint4*)(As + arow * 40 + 16 + offA);

    const int gB = (lane >= 16) ? 8 : 0;
    #pragma unroll
    for (int jj = 0; jj < 4; ++jj) {
      const int bn = ns * 64 + jj * 16 + (lane & 15);
      FragU fb;
      fb.u[0] = *(const uint4*)(Bs + bn * 40 + gB);
      fb.u[1] = *(const uint4*)(Bs + bn * 40 + gB + 16);
      acc[jj] = __builtin_amdgcn_wmma_f32_16x16x32_bf16(
          false, fa.v, false, fb.v, (short)0, acc[jj], false, false);
    }
    __syncthreads();
  }

  // ---- epilogue: VGPR r -> M = r + 8*(lane/16), N = lane%16 ----
  const int mloc = mblk + mrow * 16 + ((lane >= 16) ? 8 : 0);
  const int nl = lane & 15;
  #pragma unroll
  for (int jj = 0; jj < 4; ++jj) {
    const int n = nblk + ns * 64 + jj * 16 + nl;
    if (n >= N) continue;
    #pragma unroll
    for (int r = 0; r < 8; ++r) {
      const int m = mloc + r;
      if (m >= M) continue;
      const long long ci = (long long)m * ldc + n;
      float v = alpha * acc[jj][r];
      if (beta) v += Cb[ci];
      if (bias) v += bias[m];
      if (D)    v += D[ci];
      if (act == 1)      v = fmaxf(v, 0.0f);
      else if (act == 2) v = gelu_exact(v);
      Cb[ci] = v;
    }
  }
}

// ---------------------------------------------------------------------------
// InstanceNorm2d (population var, eps=1e-6); src may alias dst.
// ---------------------------------------------------------------------------
__global__ __launch_bounds__(256) void inorm_k(
    const float* __restrict__ src, float* __restrict__ dst,
    const float* __restrict__ w, const float* __restrict__ b, int S)
{
  const int c = blockIdx.x, tid = threadIdx.x;
  const float* x = src + (long long)c * S;
  double s = 0.0, s2 = 0.0;
  for (int i = tid; i < S; i += 256) { float v = x[i]; s += v; s2 += (double)v * v; }
  __shared__ double rs[256], rs2[256];
  rs[tid] = s; rs2[tid] = s2; __syncthreads();
  for (int o = 128; o > 0; o >>= 1) {
    if (tid < o) { rs[tid] += rs[tid + o]; rs2[tid] += rs2[tid + o]; }
    __syncthreads();
  }
  const float mean = (float)(rs[0] / S);
  const float var  = (float)(rs2[0] / S) - mean * mean;
  const float inv  = rsqrtf(var + 1e-6f);
  const float ww = w[c], bb = b[c];
  float* y = dst + (long long)c * S;
  for (int i = tid; i < S; i += 256) y[i] = (x[i] - mean) * inv * ww + bb;
}

// Tiny-K (CIN=2) 1x1 conv: out[o,s] = act(w[o*wld+woff]*x0 + w[..+1]*x1 + bias)
__global__ __launch_bounds__(256) void conv2ch_k(
    const float* __restrict__ x, const float* __restrict__ w,
    const float* __restrict__ bias, float* __restrict__ out,
    int OC, int S, int wld, int woff, int act)
{
  const long long i = (long long)blockIdx.x * 256 + threadIdx.x;
  if (i >= (long long)OC * S) return;
  const int o = (int)(i / S), s = (int)(i % S);
  float v = w[o * wld + woff] * x[s] + w[o * wld + woff + 1] * x[S + s];
  if (bias) v += bias[o];
  if (act == 2) v = gelu_exact(v);
  out[(long long)o * S + s] = v;
}

// Final decoder projection E=128 -> COUT=2
__global__ __launch_bounds__(256) void dec_final_k(
    const float* __restrict__ h, const float* __restrict__ w,
    float* __restrict__ out, int S)
{
  const int s = blockIdx.x * 256 + threadIdx.x;
  if (s >= S) return;
  float a0 = 0.f, a1 = 0.f;
  for (int e = 0; e < 128; ++e) {
    const float v = h[(long long)e * S + s];
    a0 += w[e] * v;
    a1 += w[128 + e] * v;
  }
  out[s] = a0;
  out[S + s] = a1;
}

// ---------------------------------------------------------------------------
// DFT matrix generation (ortho norm folded in); kw dimension padded to 184.
// ---------------------------------------------------------------------------
__global__ __launch_bounds__(256) void gen_dftW_k(float* __restrict__ Cc,
                                                  float* __restrict__ Sn)
{ // forward rDFT along W: [w=360][kwPad=184], exp(-2pi i w kw/360)/sqrt(360)
  const int i = blockIdx.x * 256 + threadIdx.x;
  if (i >= 360 * 184) return;
  const int w = i / 184, kw = i % 184;
  if (kw >= 181) { Cc[i] = 0.f; Sn[i] = 0.f; return; }
  const int md = (w * kw) % 360;
  const float a = -6.283185307179586f * (float)md / 360.0f;
  const float sc = rsqrtf(360.0f);
  Cc[i] = cosf(a) * sc;
  Sn[i] = sinf(a) * sc;
}

__global__ __launch_bounds__(256) void gen_dftH_k(float* __restrict__ Fr, float* __restrict__ Fi,
                                                  float* __restrict__ Gr, float* __restrict__ Gi)
{ // [181][pad 184]; F: exp(-i..)/sqrt(181);  G: exp(+i..)/sqrt(181)
  const int i = blockIdx.x * 256 + threadIdx.x;
  if (i >= 181 * 184) return;
  const int r = i / 184, c = i % 184;
  if (c >= 181) { Fr[i] = Fi[i] = Gr[i] = Gi[i] = 0.f; return; }
  const int md = (r * c) % 181;
  const float a = 6.283185307179586f * (float)md / 181.0f;
  const float sc = rsqrtf(181.0f);
  const float cv = cosf(a) * sc, sv = sinf(a) * sc;
  Fr[i] = cv; Fi[i] = -sv;
  Gr[i] = cv; Gi[i] = sv;
}

__global__ __launch_bounds__(256) void gen_idftW_k(float* __restrict__ Cc,
                                                   float* __restrict__ Sn)
{ // inverse rDFT along W, Hermitian-folded: [kw=181][w=360]
  const int i = blockIdx.x * 256 + threadIdx.x;
  if (i >= 181 * 360) return;
  const int kw = i / 360, w = i % 360;
  const int md = (kw * w) % 360;
  const float a = 6.283185307179586f * (float)md / 360.0f;
  const float ac = (kw == 0 || kw == 180) ? 1.0f : 2.0f; // Nyquist weight 1
  const float sc = rsqrtf(360.0f);
  Cc[i] =  ac * cosf(a) * sc;
  Sn[i] = -ac * sinf(a) * sc;
}

// ---------------------------------------------------------------------------
// Host side
// ---------------------------------------------------------------------------
static void gemm(hipStream_t st, const float* A, const float* B, float* C,
                 const float* D, const float* bias,
                 int M, int N, int K, int lda, int ldb, int ldc,
                 long long asb, long long bsb, long long csb, int batch,
                 int transA, float alpha, int beta, int act)
{
  dim3 g((N + 127) / 128, (M + 63) / 64, batch);
  gemm_bf16_k<<<g, dim3(256), 0, st>>>(A, B, C, D, bias, M, N, K, lda, ldb, ldc,
                                       asb, bsb, csb, transA, alpha, beta, act);
}

extern "C" void kernel_launch(void* const* d_in, const int* in_sizes, int n_in,
                              void* d_out, int out_size, void* d_ws, size_t ws_size,
                              hipStream_t stream)
{
  (void)in_sizes; (void)n_in; (void)out_size; (void)ws_size;
  const int Hh = 181, E = 128, HID = 256, Lnum = 4;
  const int S   = 181 * 360;          // 65160 spatial
  const int Wfp = 184;                // padded kw stride (181 -> 184)
  const int FP  = Hh * Wfp;           // 33304 padded freq plane per channel
  const int CH  = E * Hh;             // 23168 rows for W-direction DFT GEMMs
  const long long FB = FP;            // per-channel batch stride

  const float* x         = (const float*)d_in[0];
  const float* enc_w0    = (const float*)d_in[1];
  const float* enc_b0    = (const float*)d_in[2];
  const float* enc_w1    = (const float*)d_in[3];
  const float* pos_embed = (const float*)d_in[4];
  const float* n0w       = (const float*)d_in[5];
  const float* n0b       = (const float*)d_in[6];
  const float* w0        = (const float*)d_in[7];   // [L,2,E,HID]
  const float* w1        = (const float*)d_in[8];   // [L,2,HID,HID]
  const float* wout      = (const float*)d_in[9];   // [L,2,HID,E]
  const float* isw       = (const float*)d_in[10];  // [L,E,E]
  const float* isb       = (const float*)d_in[11];
  const float* n1w       = (const float*)d_in[12];
  const float* n1b       = (const float*)d_in[13];
  const float* fc1w      = (const float*)d_in[14];  // [L,HID,E]
  const float* fc1b      = (const float*)d_in[15];
  const float* fc2w      = (const float*)d_in[16];  // [L,E,HID]
  const float* fc2b      = (const float*)d_in[17];
  const float* dec_w0    = (const float*)d_in[18];  // [E, E+2] (lda=130 -> slow path)
  const float* dec_b0    = (const float*)d_in[19];
  const float* dec_w1    = (const float*)d_in[20];  // [2, E]
  float* out = (float*)d_out;

  size_t off = 0;
  char* base = (char*)d_ws;
  auto alloc = [&](size_t elems) -> float* {
    float* p = (float*)(base + off);
    off += ((elems * 4 + 255) / 256) * 256;
    return p;
  };
  float* h    = alloc((size_t)E * S);
  float* xn   = alloc((size_t)E * S);
  float* yb   = alloc((size_t)E * S);
  float* mlph = alloc((size_t)HID * S);
  float* P1 = alloc((size_t)E * FP);   // Z1r / Yr
  float* P2 = alloc((size_t)E * FP);   // Z1i / Yi
  float* P3 = alloc((size_t)E * FP);   // Zr  / Y1r
  float* P4 = alloc((size_t)E * FP);   // Zi  / Y1i
  float* Q1 = alloc((size_t)HID * FP); // Hr
  float* Q2 = alloc((size_t)HID * FP); // Hi
  float* Q3 = alloc((size_t)HID * FP); // H2r
  float* Q4 = alloc((size_t)HID * FP); // H2i
  float* FwC = alloc((size_t)360 * Wfp);
  float* FwS = alloc((size_t)360 * Wfp);
  float* FHr = alloc((size_t)181 * Wfp);
  float* FHi = alloc((size_t)181 * Wfp);
  float* GHr = alloc((size_t)181 * Wfp);
  float* GHi = alloc((size_t)181 * Wfp);
  float* ICw = alloc((size_t)181 * 360);
  float* ISw = alloc((size_t)181 * 360);

  gen_dftW_k<<<(360 * Wfp + 255) / 256, 256, 0, stream>>>(FwC, FwS);
  gen_dftH_k<<<(181 * Wfp + 255) / 256, 256, 0, stream>>>(FHr, FHi, GHr, GHi);
  gen_idftW_k<<<(181 * 360 + 255) / 256, 256, 0, stream>>>(ICw, ISw);

  // encoder: h = enc_w1 * gelu(enc_w0*x + b0) + pos_embed
  conv2ch_k<<<(int)(((long long)E * S + 255) / 256), 256, 0, stream>>>(
      x, enc_w0, enc_b0, yb, E, S, 2, 0, 2);
  gemm(stream, enc_w1, yb, h, pos_embed, nullptr, E, S, E, E, S, S, 0, 0, 0, 1,
       0, 1.0f, 0, 0);

  for (int l = 0; l < Lnum; ++l) {
    const float* Wr  = w0   + (size_t)l * 2 * E * HID;
    const float* Wi  = Wr   + (size_t)E * HID;
    const float* W1r = w1   + (size_t)l * 2 * HID * HID;
    const float* W1i = W1r  + (size_t)HID * HID;
    const float* Wor = wout + (size_t)l * 2 * HID * E;
    const float* Woi = Wor  + (size_t)HID * E;
    const float* iswl = isw + (size_t)l * E * E;
    const float* isbl = isb + (size_t)l * E;
    const float* f1w = fc1w + (size_t)l * HID * E;
    const float* f1b = fc1b + (size_t)l * HID;
    const float* f2w = fc2w + (size_t)l * E * HID;
    const float* f2b = fc2b + (size_t)l * E;

    inorm_k<<<E, 256, 0, stream>>>(h, xn, n0w + l * E, n0b + l * E, S);

    // rfft along W: Z1 = xn * Fw   [CH x 360] x [360 x 181(pad184)]
    gemm(stream, xn, FwC, P1, nullptr, nullptr, CH, 181, 360, 360, Wfp, Wfp,
         0, 0, 0, 1, 0, 1.0f, 0, 0);
    gemm(stream, xn, FwS, P2, nullptr, nullptr, CH, 181, 360, 360, Wfp, Wfp,
         0, 0, 0, 1, 0, 1.0f, 0, 0);
    // DFT along H (complex), batched over channels: Z = FH * Z1
    gemm(stream, FHr, P1, P3, nullptr, nullptr, 181, 181, 181, Wfp, Wfp, Wfp,
         0, FB, FB, E, 0, 1.0f, 0, 0);
    gemm(stream, FHi, P2, P3, nullptr, nullptr, 181, 181, 181, Wfp, Wfp, Wfp,
         0, FB, FB, E, 0, -1.0f, 1, 0);
    gemm(stream, FHr, P2, P4, nullptr, nullptr, 181, 181, 181, Wfp, Wfp, Wfp,
         0, FB, FB, E, 0, 1.0f, 0, 0);
    gemm(stream, FHi, P1, P4, nullptr, nullptr, 181, 181, 181, Wfp, Wfp, Wfp,
         0, FB, FB, E, 0, 1.0f, 1, 0);

    // spectral layer 0: H = crelu(W0^T * Z)   (pad columns = harmless garbage)
    gemm(stream, Wr, P3, Q1, nullptr, nullptr, HID, FP, E, HID, FP, FP, 0, 0, 0, 1,
         1, 1.0f, 0, 0);
    gemm(stream, Wi, P4, Q1, nullptr, nullptr, HID, FP, E, HID, FP, FP, 0, 0, 0, 1,
         1, -1.0f, 1, 1);
    gemm(stream, Wr, P4, Q2, nullptr, nullptr, HID, FP, E, HID, FP, FP, 0, 0, 0, 1,
         1, 1.0f, 0, 0);
    gemm(stream, Wi, P3, Q2, nullptr, nullptr, HID, FP, E, HID, FP, FP, 0, 0, 0, 1,
         1, 1.0f, 1, 0);
    // layer 1
    gemm(stream, W1r, Q1, Q3, nullptr, nullptr, HID, FP, HID, HID, FP, FP, 0, 0, 0, 1,
         1, 1.0f, 0, 0);
    gemm(stream, W1i, Q2, Q3, nullptr, nullptr, HID, FP, HID, HID, FP, FP, 0, 0, 0, 1,
         1, -1.0f, 1, 1);
    gemm(stream, W1r, Q2, Q4, nullptr, nullptr, HID, FP, HID, HID, FP, FP, 0, 0, 0, 1,
         1, 1.0f, 0, 0);
    gemm(stream, W1i, Q1, Q4, nullptr, nullptr, HID, FP, HID, HID, FP, FP, 0, 0, 0, 1,
         1, 1.0f, 1, 0);
    // wout
    gemm(stream, Wor, Q3, P1, nullptr, nullptr, E, FP, HID, E, FP, FP, 0, 0, 0, 1,
         1, 1.0f, 0, 0);
    gemm(stream, Woi, Q4, P1, nullptr, nullptr, E, FP, HID, E, FP, FP, 0, 0, 0, 1,
         1, -1.0f, 1, 0);
    gemm(stream, Wor, Q4, P2, nullptr, nullptr, E, FP, HID, E, FP, FP, 0, 0, 0, 1,
         1, 1.0f, 0, 0);
    gemm(stream, Woi, Q3, P2, nullptr, nullptr, E, FP, HID, E, FP, FP, 0, 0, 0, 1,
         1, 1.0f, 1, 0);

    // inverse DFT along H: Y1 = GH * Y (batched)
    gemm(stream, GHr, P1, P3, nullptr, nullptr, 181, 181, 181, Wfp, Wfp, Wfp,
         0, FB, FB, E, 0, 1.0f, 0, 0);
    gemm(stream, GHi, P2, P3, nullptr, nullptr, 181, 181, 181, Wfp, Wfp, Wfp,
         0, FB, FB, E, 0, -1.0f, 1, 0);
    gemm(stream, GHr, P2, P4, nullptr, nullptr, 181, 181, 181, Wfp, Wfp, Wfp,
         0, FB, FB, E, 0, 1.0f, 0, 0);
    gemm(stream, GHi, P1, P4, nullptr, nullptr, 181, 181, 181, Wfp, Wfp, Wfp,
         0, FB, FB, E, 0, 1.0f, 1, 0);
    // inverse rDFT along W: y = Y1r*ICw + Y1i*ISw
    gemm(stream, P3, ICw, yb, nullptr, nullptr, CH, 360, 181, Wfp, 360, 360,
         0, 0, 0, 1, 0, 1.0f, 0, 0);
    gemm(stream, P4, ISw, yb, nullptr, nullptr, CH, 360, 181, Wfp, 360, 360,
         0, 0, 0, 1, 0, 1.0f, 1, 0);

    // inner skip: y += isw*xn + isb
    gemm(stream, iswl, xn, yb, nullptr, isbl, E, S, E, E, S, S, 0, 0, 0, 1,
         0, 1.0f, 1, 0);
    inorm_k<<<E, 256, 0, stream>>>(yb, yb, n1w + l * E, n1b + l * E, S);
    // MLP: h_next = fc2*gelu(fc1*y + b1) + b2 + xn
    gemm(stream, f1w, yb, mlph, nullptr, f1b, HID, S, E, E, S, S, 0, 0, 0, 1,
         0, 1.0f, 0, 2);
    gemm(stream, f2w, mlph, h, xn, f2b, E, S, HID, HID, S, S, 0, 0, 0, 1,
         0, 1.0f, 0, 0);
  }

  // decoder: gelu(dec_w0 * [h; x] + b) then final E->2
  conv2ch_k<<<(int)(((long long)E * S + 255) / 256), 256, 0, stream>>>(
      x, dec_w0, nullptr, yb, E, S, E + 2, E, 0);
  gemm(stream, dec_w0, h, yb, nullptr, dec_b0, E, S, E, E + 2, S, S, 0, 0, 0, 1,
       0, 1.0f, 1, 2);
  dec_final_k<<<(S + 255) / 256, 256, 0, stream>>>(yb, dec_w1, out, S);
}
